// InfiniAttentionMemory_73710228734429
// MI455X (gfx1250) — matrix operand order repaired and checked by reading, not compiled, not run on hardware
//
#include <hip/hip_runtime.h>

// ---------------------------------------------------------------------------
// CDNA5 / gfx1250 Infini-attention memory kernel (wave32, WMMA bf16)
// ---------------------------------------------------------------------------
typedef float v8f  __attribute__((ext_vector_type(8)));
typedef __bf16 v16bf __attribute__((ext_vector_type(16)));

union Frag16 {                 // 16 bf16 = 8 VGPRs, matches WMMA A/B operand
    v16bf        v;
    unsigned int u[8];
};

// pack two f32 -> packed bf16 pair in ONE v_perm_b32:
//   result[15:0]  = a[31:16]  (bf16 truncate)
//   result[31:16] = b[31:16]
__device__ __forceinline__ unsigned int pk2(float a, float b) {
    union { float f; unsigned int i; } x, y; x.f = a; y.f = b;
    // v_perm_b32: sel bytes 0-3 pick from S1 (=x), 4-7 from S0 (=y)
    return __builtin_amdgcn_perm(y.i, x.i, 0x07060302u);
}

__device__ __forceinline__ v8f wmma_bf16(const Frag16& a, const Frag16& b, v8f c) {
    // D = A(16x32 bf16) * B(32x16 bf16) + C(16x16 f32)
    return __builtin_amdgcn_wmma_f32_16x16x32_bf16(
        /*neg_a=*/false, a.v, /*neg_b=*/false, b.v,
        /*c_mod=*/(short)0, c, /*reuse_a=*/false, /*reuse_b=*/false);
}

__device__ __forceinline__ v8f vzero8() {
    v8f z = {0.f, 0.f, 0.f, 0.f, 0.f, 0.f, 0.f, 0.f};
    return z;
}

// ---------------------------------------------------------------------------
// Kernel 0: seed M_new = M, z_new = z (update kernel atomically accumulates)
// ---------------------------------------------------------------------------
__global__ void k_init(const float* __restrict__ M, const float* __restrict__ z,
                       float* __restrict__ outM, float* __restrict__ outz) {
    int i = blockIdx.x * 256 + threadIdx.x;
    if (i < 64 * 64)            outM[i] = M[i];
    else if (i < 64 * 64 + 64)  outz[i - 64 * 64] = z[i - 64 * 64];
}

// ---------------------------------------------------------------------------
// Kernel 1: retrieve + gate.
//   per wave: 16-token tile; GEMM1 sigmaQ = elu(Q Wq^T)+1 (8 WMMA),
//   LDS transpose, denominator, GEMM2 num = sigmaQ @ M (8 WMMA), gated store.
// ---------------------------------------------------------------------------
__global__ __launch_bounds__(256) void
k_retrieve(const float* __restrict__ q, const float* __restrict__ local_out,
           const float* __restrict__ wq, const float* __restrict__ mmat,
           const float* __restrict__ zvec, const float* __restrict__ beta,
           float* __restrict__ out, int ntok) {
    // B-fragment layout: frag f=(ntile*2+chunk); lane l: n=l&15, h=l>>4;
    //   element j (bf16): Bmat[K = chunk*32 + 8*h + j + (j >= 8 ? 8 : 0)][ntile*16 + n]
    __shared__ unsigned int sWq[8][32][8];   // Wq^T frags (B of GEMM1)
    __shared__ unsigned int sM [8][32][8];   // M    frags (B of GEMM2)
    __shared__ float        sZ[64];
    __shared__ float        sQ[8][16][64];   // per-wave sigmaQ staging
    __shared__ float        sDen[8][16];

    const int tid  = threadIdx.x;
    const int lane = tid & 31;
    const int w    = tid >> 5;
    const int n    = lane & 15;
    const int h    = lane >> 4;

    // ---- stage weight fragments (256 threads -> 8 frags x 32 lanes) ----
    {
        const int f  = w;           // fragment id 0..7
        const int nt = f >> 1, c = f & 1;
        const int nn = lane & 15, hh = lane >> 4;
        #pragma unroll
        for (int j = 0; j < 8; ++j) {
            const int k0 = c * 32 + 8 * hh + 2 * j + (j >= 4 ? 8 : 0);
            // B of GEMM1 = Wq^T : B[d][k] = Wq[k*64+d]  (contiguous in d)
            float a0 = wq[(nt * 16 + nn) * 64 + k0];
            float a1 = wq[(nt * 16 + nn) * 64 + k0 + 1];
            sWq[f][lane][j] = pk2(a0, a1);
            // B of GEMM2 = M : B[k][v] = M[k*64+v]      (stride 64 in k)
            float b0 = mmat[(k0    ) * 64 + nt * 16 + nn];
            float b1 = mmat[(k0 + 1) * 64 + nt * 16 + nn];
            sM[f][lane][j] = pk2(b0, b1);
        }
    }
    if (tid < 64) sZ[tid] = zvec[tid];
    __syncthreads();

    const float g  = 1.0f / (1.0f + __expf(-beta[0]));
    const int tile = blockIdx.x * 8 + w;
    if (tile * 16 >= ntok) return;

    const long tok0  = (long)tile * 16;
    const long token = tok0 + n;          // this lane's A-matrix row

    // ---- A fragments from Q (16-bit A 16x32 layout) ----
    Frag16 aQ[2];
    #pragma unroll
    for (int c = 0; c < 2; ++c) {
        const float* p = q + token * 64 + c * 32 + 8 * h;
        float4 q0 = ((const float4*)p)[0];
        float4 q1 = ((const float4*)p)[1];
        float4 q2 = ((const float4*)(p + 16))[0];
        float4 q3 = ((const float4*)(p + 16))[1];
        aQ[c].u[0] = pk2(q0.x, q0.y); aQ[c].u[1] = pk2(q0.z, q0.w);
        aQ[c].u[2] = pk2(q1.x, q1.y); aQ[c].u[3] = pk2(q1.z, q1.w);
        aQ[c].u[4] = pk2(q2.x, q2.y); aQ[c].u[5] = pk2(q2.z, q2.w);
        aQ[c].u[6] = pk2(q3.x, q3.y); aQ[c].u[7] = pk2(q3.z, q3.w);
    }

    // ---- GEMM1: sigmaQ_raw = Q @ Wq^T ----
    v8f acc[4];
    #pragma unroll
    for (int t = 0; t < 4; ++t) acc[t] = vzero8();
    #pragma unroll
    for (int t = 0; t < 4; ++t) {
        #pragma unroll
        for (int c = 0; c < 2; ++c) {
            Frag16 b;
            ((uint4*)b.u)[0] = ((const uint4*)&sWq[t * 2 + c][lane][0])[0];
            ((uint4*)b.u)[1] = ((const uint4*)&sWq[t * 2 + c][lane][0])[1];
            acc[t] = wmma_bf16(aQ[c], b, acc[t]);
        }
    }

    // ---- sigma = elu+1, stage [16 tok x 64 k] in LDS ----
    #pragma unroll
    for (int t = 0; t < 4; ++t) {
        #pragma unroll
        for (int r = 0; r < 8; ++r) {
            float x = acc[t][r];
            float s = x > 0.0f ? x + 1.0f : __expf(x);
            sQ[w][r + 8 * h][t * 16 + n] = s;   // (m = r+8h, k = t*16+n)
        }
    }
    asm volatile("s_wait_dscnt 0" ::: "memory");

    // ---- denominator: den[t] = sigmaQ[t,:] . z ----
    float part = 0.0f;
    #pragma unroll
    for (int k = 0; k < 32; ++k) part += sQ[w][n][32 * h + k] * sZ[32 * h + k];
    part += __shfl_xor(part, 16, 32);
    if (lane < 16) sDen[w][lane] = 1.0f / (part + 1e-8f);
    asm volatile("s_wait_dscnt 0" ::: "memory");

    // ---- A' fragments (sigmaQ as A-matrix) from LDS ----
    Frag16 aS[2];
    #pragma unroll
    for (int c = 0; c < 2; ++c) {
        const float* p = &sQ[w][n][c * 32 + 8 * h];
        float4 s0 = ((const float4*)p)[0];
        float4 s1 = ((const float4*)p)[1];
        float4 s2 = ((const float4*)(p + 16))[0];
        float4 s3 = ((const float4*)(p + 16))[1];
        aS[c].u[0] = pk2(s0.x, s0.y); aS[c].u[1] = pk2(s0.z, s0.w);
        aS[c].u[2] = pk2(s1.x, s1.y); aS[c].u[3] = pk2(s1.z, s1.w);
        aS[c].u[4] = pk2(s2.x, s2.y); aS[c].u[5] = pk2(s2.z, s2.w);
        aS[c].u[6] = pk2(s3.x, s3.y); aS[c].u[7] = pk2(s3.z, s3.w);
    }

    // ---- GEMM2: numerator = sigmaQ @ M ----
    v8f acc2[4];
    #pragma unroll
    for (int t = 0; t < 4; ++t) acc2[t] = vzero8();
    #pragma unroll
    for (int t = 0; t < 4; ++t) {
        #pragma unroll
        for (int c = 0; c < 2; ++c) {
            Frag16 b;
            ((uint4*)b.u)[0] = ((const uint4*)&sM[t * 2 + c][lane][0])[0];
            ((uint4*)b.u)[1] = ((const uint4*)&sM[t * 2 + c][lane][0])[1];
            acc2[t] = wmma_bf16(aS[c], b, acc2[t]);
        }
    }

    float rd[8];
    #pragma unroll
    for (int r = 0; r < 8; ++r) rd[r] = sDen[w][r + 8 * h];

    // ---- gate + store: out = g*A_mem + (1-g)*local ----
    #pragma unroll
    for (int t = 0; t < 4; ++t) {
        #pragma unroll
        for (int r = 0; r < 8; ++r) {
            long idx  = (tok0 + r + 8 * h) * 64 + t * 16 + n;
            float am  = acc2[t][r] * rd[r];
            out[idx]  = g * am + (1.0f - g) * local_out[idx];
        }
    }
}

// ---------------------------------------------------------------------------
// Kernel 2: update.  Wave owns k-tile kt (16 rows of M) and 256 tokens.
//   Per 32-token slab: 4 WMMA (sigmaK = elu(K Wk^T)+1), LDS transpose,
//   4 WMMA accumulating sigmaK^T @ V (K-dim = 32 tokens per WMMA).
//   Partials -> global_atomic_add_f32 into M_new / z_new.
// ---------------------------------------------------------------------------
__global__ __launch_bounds__(256) void
k_update(const float* __restrict__ keys, const float* __restrict__ vals,
         const float* __restrict__ wk,
         float* __restrict__ outM, float* __restrict__ outz, int ntok) {
    __shared__ unsigned int sWk[2][32][8];   // Wk^T frags for this kt
    __shared__ float        sK[8][32][16];   // per-wave sigmaK slab staging

    const int tid  = threadIdx.x;
    const int lane = tid & 31;
    const int w    = tid >> 5;
    const int n    = lane & 15;
    const int h    = lane >> 4;
    const int kt   = blockIdx.x & 3;          // which 16 rows of M
    const int red  = blockIdx.x >> 2;         // reducer index

    if (tid < 64) {
        const int c = tid >> 5, l = tid & 31;
        const int nn = l & 15, hh = l >> 4;
        #pragma unroll
        for (int j = 0; j < 8; ++j) {
            const int k0 = c * 32 + 8 * hh + 2 * j + (j >= 4 ? 8 : 0);
            float a0 = wk[(kt * 16 + nn) * 64 + k0];
            float a1 = wk[(kt * 16 + nn) * 64 + k0 + 1];
            sWk[c][l][j] = pk2(a0, a1);
        }
    }
    __syncthreads();

    const long base = ((long)red * 8 + w) * 256;   // first token of this wave
    if (base >= ntok) return;

    v8f accV[4];
    #pragma unroll
    for (int vt = 0; vt < 4; ++vt) accV[vt] = vzero8();
    float zacc = 0.0f;

    for (int s8 = 0; s8 < 8; ++s8) {
        const long tok0 = base + s8 * 32;
        if (s8 + 1 < 8) {  // software prefetch of next slab (global_prefetch_b8)
            __builtin_prefetch(keys + (tok0 + 32 + lane) * 64, 0, 1);
            __builtin_prefetch(vals + (tok0 + 32 + lane) * 64, 0, 1);
        }

        // ---- A frags: K slab, two 16-token row tiles ----
        Frag16 aK[2][2];
        #pragma unroll
        for (int mt = 0; mt < 2; ++mt) {
            #pragma unroll
            for (int c = 0; c < 2; ++c) {
                const float* p = keys + (tok0 + mt * 16 + n) * 64 + c * 32 + 8 * h;
                float4 q0 = ((const float4*)p)[0];
                float4 q1 = ((const float4*)p)[1];
                float4 q2 = ((const float4*)(p + 16))[0];
                float4 q3 = ((const float4*)(p + 16))[1];
                aK[mt][c].u[0] = pk2(q0.x, q0.y); aK[mt][c].u[1] = pk2(q0.z, q0.w);
                aK[mt][c].u[2] = pk2(q1.x, q1.y); aK[mt][c].u[3] = pk2(q1.z, q1.w);
                aK[mt][c].u[4] = pk2(q2.x, q2.y); aK[mt][c].u[5] = pk2(q2.z, q2.w);
                aK[mt][c].u[6] = pk2(q3.x, q3.y); aK[mt][c].u[7] = pk2(q3.z, q3.w);
            }
        }

        // ---- sigmaK slab = K_slab @ Wk^T (this kt's 16 columns) ----
        v8f ck[2] = {vzero8(), vzero8()};
        #pragma unroll
        for (int c = 0; c < 2; ++c) {
            Frag16 b;
            ((uint4*)b.u)[0] = ((const uint4*)&sWk[c][lane][0])[0];
            ((uint4*)b.u)[1] = ((const uint4*)&sWk[c][lane][0])[1];
            ck[0] = wmma_bf16(aK[0][c], b, ck[0]);
            ck[1] = wmma_bf16(aK[1][c], b, ck[1]);
        }

        // ---- elu+1, accumulate z, stage [32 tok x 16 k] in LDS ----
        #pragma unroll
        for (int mt = 0; mt < 2; ++mt) {
            #pragma unroll
            for (int r = 0; r < 8; ++r) {
                float x = ck[mt][r];
                float s = x > 0.0f ? x + 1.0f : __expf(x);
                zacc += s;
                sK[w][mt * 16 + r + 8 * h][n] = s;
            }
        }
        asm volatile("s_wait_dscnt 0" ::: "memory");

        // ---- A' = sigmaK^T [16 k x 32 tok] (transpose via LDS) ----
        Frag16 aT;
        #pragma unroll
        for (int j = 0; j < 4; ++j) {
            aT.u[j]     = pk2(sK[w][     8 * h + 2 * j][n], sK[w][     8 * h + 2 * j + 1][n]);
            aT.u[4 + j] = pk2(sK[w][16 + 8 * h + 2 * j][n], sK[w][16 + 8 * h + 2 * j + 1][n]);
        }

        // ---- accumulate sigmaK^T @ V slab ----
        #pragma unroll
        for (int vt = 0; vt < 4; ++vt) {
            Frag16 bv;
            #pragma unroll
            for (int jj = 0; jj < 8; ++jj) {
                const long t0 = tok0 + 8 * h + 2 * jj + (jj >= 4 ? 8 : 0);
                float f0 = vals[t0 * 64 + vt * 16 + n];
                float f1 = vals[(t0 + 1) * 64 + vt * 16 + n];
                bv.u[jj] = pk2(f0, f1);
            }
            accV[vt] = wmma_bf16(aT, bv, accV[vt]);
        }
    }

    // ---- commit partials ----
    float zl = zacc + __shfl_xor(zacc, 16, 32);
    if (lane < 16) unsafeAtomicAdd(&outz[kt * 16 + lane], zl);
    #pragma unroll
    for (int vt = 0; vt < 4; ++vt) {
        #pragma unroll
        for (int r = 0; r < 8; ++r) {
            unsafeAtomicAdd(&outM[(kt * 16 + r + 8 * h) * 64 + vt * 16 + n],
                            accV[vt][r]);
        }
    }
}

// ---------------------------------------------------------------------------
// Launch
// ---------------------------------------------------------------------------
extern "C" void kernel_launch(void* const* d_in, const int* in_sizes, int n_in,
                              void* d_out, int out_size, void* d_ws, size_t ws_size,
                              hipStream_t stream) {
    const float* q     = (const float*)d_in[0];
    const float* keys  = (const float*)d_in[1];
    const float* vals  = (const float*)d_in[2];
    const float* local = (const float*)d_in[3];
    const float* wq    = (const float*)d_in[4];
    const float* wk    = (const float*)d_in[5];
    const float* beta  = (const float*)d_in[6];
    const float* M     = (const float*)d_in[7];
    const float* z     = (const float*)d_in[8];

    const int ntok = in_sizes[0] / 64;            // B*S = 131072
    float* out  = (float*)d_out;
    float* outM = out + (size_t)ntok * 64;
    float* outz = outM + 64 * 64;

    // 0) seed M_new / z_new
    k_init<<<(64 * 64 + 64 + 255) / 256, 256, 0, stream>>>(M, z, outM, outz);

    // 1) retrieve + gate: 1 wave per 16-token tile, 8 waves per block
    const int ntiles = ntok / 16;
    k_retrieve<<<(ntiles + 7) / 8, 256, 0, stream>>>(q, local, wq, M, z, beta,
                                                     out, ntok);

    // 2) update: 4 k-tiles x R reducers; each wave covers 256 tokens
    const int R = (ntok + 2047) / 2048;
    k_update<<<R * 4, 256, 0, stream>>>(keys, vals, wk, outM, outz, ntok);
}
